// DGMG_18726057410999
// MI455X (gfx1250) — compile-verified, matrix operand order checked by reading, not run on hardware
//
#include <hip/hip_runtime.h>
#include <math.h>

// ---------- types ----------
typedef __attribute__((ext_vector_type(16))) __bf16 v16bf;
typedef __attribute__((ext_vector_type(8)))  float  v8f;
typedef unsigned short u16;
typedef unsigned int   u32;

#define BB 128      // batch
#define NN 32       // nodes
#define DD 256
#define LDT 40      // LDS row stride (u16 units): keeps 16B alignment, staggers banks

// fp32 -> packed 2x bf16 via native __bf16 conversion (RNE, legalized by clang)
__device__ __forceinline__ u32 pack_bf16(float x, float y) {
    union { __bf16 h[2]; u32 u; } p;
    p.h[0] = (__bf16)x;
    p.h[1] = (__bf16)y;
    return p.u;
}

__device__ __forceinline__ u16 f2bf(float f) {
    union { __bf16 h[2]; u16 s[2]; } p;
    p.h[0] = (__bf16)f;
    return p.s[0];
}

// ---------- weight transpose + fp32->bf16 : Wt[n*K+k] = bf16(W[k*N+n]) ----------
__global__ void convT_kernel(const float* __restrict__ W, u16* __restrict__ Wt, int K, int N) {
    int k = blockIdx.x * 16 + threadIdx.x;
    int n = blockIdx.y * 16 + threadIdx.y;
    if (k < K && n < N) Wt[(size_t)n * K + k] = f2bf(W[(size_t)k * N + n]);
}

// ---------- generic WMMA GEMM: C[M x N] = act(Agather[M x K] @ W + bias*bscale) ----------
// Wt bf16 transposed (N x K row-major). C fp32 with ldc.
// amode: 0 = plain A (row-major, lda=K)
//        1 = feat gather: row r -> b=r/vv, j=r%vv; cols [0,256)=h[b][j], [256,512)=h[b][vv]  (A = h)
//        2 = concat:      row b;            cols [0,256)=A[b],    [256,512)=Asec[b][vv]      (A = hG, Asec = h)
// act:   0 = none, 1 = sigmoid, 2 = relu then zero rows with (row&31) >= count (GCN node mask)
__global__ __launch_bounds__(256)
void gemm_wmma_kernel(const float* __restrict__ A, const float* __restrict__ Asec,
                      int amode, int vv,
                      const u16* __restrict__ Wt,
                      const float* __restrict__ bias, float bscale,
                      float* __restrict__ C, int ldc,
                      int M, int N, int K, int act, int count) {
    __shared__ u16 As[2][64 * LDT];   // ping-pong A tiles (fp32->bf16 staged)
    __shared__ u16 Bs[2][64 * LDT];   // ping-pong B tiles (async global->LDS)

    const int t     = threadIdx.x;
    const int mBase = blockIdx.x * 64;
    const int nBase = blockIdx.y * 64;
    const int wave  = t >> 5;
    const int lane  = t & 31;
    const int l     = lane & 15;
    const int half  = lane >> 4;
    const int mSub  = (wave & 3) * 16;   // 4 M-subtiles
    const int nSub  = (wave >> 2) * 32;  // 2 N-subtiles of 32 (2x16)

    v8f acc0 = {};
    v8f acc1 = {};

    const int rowA = t >> 2;         // 0..63
    const int colA = (t & 3) * 8;    // 0,8,16,24
    const int r    = mBase + rowA;   // global output row this thread stages

    // Precompute the two half-K source row pointers for the staged A row.
    const float* srcLo;   // covers k in [0,256) (or whole K for amode 0)
    const float* srcHi;   // covers k in [256,512) for the gather modes
    if (amode == 1) {
        int b = r / vv, j = r - b * vv;
        srcLo = A + ((size_t)b * NN + j)  * DD;
        srcHi = A + ((size_t)b * NN + vv) * DD;
    } else if (amode == 2) {
        srcLo = A + (size_t)r * DD;
        srcHi = Asec + ((size_t)r * NN + vv) * DD;
    } else {
        srcLo = A + (size_t)r * K;
        srcHi = srcLo;   // unused (plain rows are K-contiguous)
    }
    const bool plain = (amode == 0);
    const u16* wrow  = Wt + (size_t)(nBase + rowA) * K + colA;

    // ---- staging helpers ----
    auto stageA = [&](int k0, int buf) {
        int kk = k0 + colA;
        const float* ap = (plain || kk < 256) ? (srcLo + kk) : (srcHi + (kk - 256));
        float4 f0 = ((const float4*)ap)[0];
        float4 f1 = ((const float4*)ap)[1];
        __builtin_prefetch(ap + 32, 0, 3);   // speculative WGP-scope prefetch of next K-tile
        uint4 pk;
        pk.x = pack_bf16(f0.x, f0.y);
        pk.y = pack_bf16(f0.z, f0.w);
        pk.z = pack_bf16(f1.x, f1.y);
        pk.w = pack_bf16(f1.z, f1.w);
        *(uint4*)&As[buf][rowA * LDT + colA] = pk;
    };
    auto stageB = [&](int k0, int buf) {
        const u16* bp = wrow + k0;
        u32 ldsOff = (u32)(uintptr_t)&Bs[buf][rowA * LDT + colA];  // wave-relative LDS byte addr
        asm volatile("global_load_async_to_lds_b128 %0, %1, off"
                     :: "v"(ldsOff), "v"(bp) : "memory");
    };

    // ---- prologue: stage tile 0 ----
    stageA(0, 0);
    stageB(0, 0);
    asm volatile("s_wait_asynccnt 0x0" ::: "memory");
    __syncthreads();

    const int nt = K >> 5;
    for (int i = 0; i < nt; ++i) {
        const int cur = i & 1;
        if (i + 1 < nt) {            // overlap next tile's staging with this tile's math
            stageA((i + 1) * 32, cur ^ 1);
            stageB((i + 1) * 32, cur ^ 1);
        }

        // ---- fragments per ISA 16-bit A/B layout: lane half selects K-group of 8 ----
        const u16* Ac = &As[cur][0];
        const u16* Bc = &Bs[cur][0];
        v16bf a, b0, b1;
        ((uint4*)&a)[0]  = *(const uint4*)&Ac[(mSub + l) * LDT + half * 8];
        ((uint4*)&a)[1]  = *(const uint4*)&Ac[(mSub + l) * LDT + 16 + half * 8];
        ((uint4*)&b0)[0] = *(const uint4*)&Bc[(nSub + l) * LDT + half * 8];
        ((uint4*)&b0)[1] = *(const uint4*)&Bc[(nSub + l) * LDT + 16 + half * 8];
        ((uint4*)&b1)[0] = *(const uint4*)&Bc[(nSub + 16 + l) * LDT + half * 8];
        ((uint4*)&b1)[1] = *(const uint4*)&Bc[(nSub + 16 + l) * LDT + 16 + half * 8];

        acc0 = __builtin_amdgcn_wmma_f32_16x16x32_bf16(false, a, false, b0, (short)0, acc0, false, false);
        acc1 = __builtin_amdgcn_wmma_f32_16x16x32_bf16(false, a, false, b1, (short)0, acc1, false, false);

        if (i + 1 < nt) {
            asm volatile("s_wait_asynccnt 0x0" ::: "memory");  // next B tile resident in LDS
            __syncthreads();                                   // + all waves' A stores visible
        }
    }

    // ---- epilogue: C layout — VGPR i holds M = i + half*8, N = l ----
    const int c0 = nBase + nSub + l;
    const int c1 = c0 + 16;
    const float bb0 = bias[c0] * bscale;
    const float bb1 = bias[c1] * bscale;
    #pragma unroll
    for (int i = 0; i < 8; ++i) {
        int rr = mBase + mSub + half * 8 + i;
        float v0 = acc0[i] + bb0;
        float v1 = acc1[i] + bb1;
        if (act == 1) {
            v0 = 1.0f / (1.0f + expf(-v0));
            v1 = 1.0f / (1.0f + expf(-v1));
        } else if (act == 2) {
            v0 = fmaxf(v0, 0.0f);
            v1 = fmaxf(v1, 0.0f);
            if ((rr & 31) >= count) { v0 = 0.0f; v1 = 0.0f; }
        }
        C[(size_t)rr * ldc + c0] = v0;
        C[(size_t)rr * ldc + c1] = v1;
    }
}

// ---------- small-N head: out[m][j] = hid[m][:] . W2[:,j] + b2[j], N2 in {1,2} ----------
__global__ void head_kernel(const float* __restrict__ hid, const float* __restrict__ W2,
                            const float* __restrict__ b2, float* __restrict__ out,
                            int M, int K, int N2) {
    int row  = blockIdx.x * (blockDim.x >> 5) + (threadIdx.x >> 5);
    int lane = threadIdx.x & 31;
    if (row >= M) return;
    for (int j = 0; j < N2; ++j) {
        float p = 0.0f;
        for (int k = lane; k < K; k += 32) p += hid[(size_t)row * K + k] * W2[(size_t)k * N2 + j];
        for (int off = 16; off > 0; off >>= 1) p += __shfl_xor(p, off);
        if (lane == 0) out[(size_t)row * N2 + j] = p + b2[j];
    }
}

// ---------- loss: -mean(log_softmax(logits,axis=1)[:, sel]), logits (128 x 2) ----------
__global__ void loss_bin2_kernel(const float* __restrict__ logits, int sel, float* loss) {
    __shared__ float red[128];
    int b = threadIdx.x;
    float l0 = logits[b * 2 + 0], l1 = logits[b * 2 + 1];
    float m = fmaxf(l0, l1);
    float lse = m + logf(expf(l0 - m) + expf(l1 - m));
    float sv = sel ? l1 : l0;
    red[b] = -(sv - lse) * (1.0f / 128.0f);
    __syncthreads();
    for (int s = 64; s > 0; s >>= 1) { if (b < s) red[b] += red[b + s]; __syncthreads(); }
    if (b == 0) atomicAdd(loss, red[0]);
}

// ---------- loss: -mean(log_sigmoid(sign * z)), z (128) ----------
__global__ void loss_logsig_kernel(const float* __restrict__ z, float sign, float* loss) {
    __shared__ float red[128];
    int b = threadIdx.x;
    float x = sign * z[b];
    float ls = (x < 0.0f) ? (x - log1pf(expf(x))) : (-log1pf(expf(-x)));
    red[b] = -ls * (1.0f / 128.0f);
    __syncthreads();
    for (int s = 64; s > 0; s >>= 1) { if (b < s) red[b] += red[b + s]; __syncthreads(); }
    if (b == 0) atomicAdd(loss, red[0]);
}

// ---------- loss: -mean_b(log_softmax(s[b,:v])[src]), s flat (B*v) ----------
__global__ void loss_rowsm_kernel(const float* __restrict__ s, int v, int src, float* loss) {
    int b    = blockIdx.x * (blockDim.x >> 5) + (threadIdx.x >> 5);
    int lane = threadIdx.x & 31;
    if (b >= BB) return;
    float m = -INFINITY;
    for (int j = lane; j < v; j += 32) m = fmaxf(m, s[(size_t)b * v + j]);
    for (int off = 16; off > 0; off >>= 1) m = fmaxf(m, __shfl_xor(m, off));
    float se = 0.0f;
    for (int j = lane; j < v; j += 32) se += expf(s[(size_t)b * v + j] - m);
    for (int off = 16; off > 0; off >>= 1) se += __shfl_xor(se, off);
    if (lane == 0) {
        float lse = m + logf(se);
        atomicAdd(loss, -(s[(size_t)b * v + src] - lse) * (1.0f / 128.0f));
    }
}

// ---------- GCN gather: chain graph, g = (A_hat/deg) h over first `count` nodes ----------
__global__ void gather_kernel(const float* __restrict__ h, float* __restrict__ g, int count) {
    int bn = blockIdx.x;           // b*32 + n
    int n  = bn & 31;
    int d  = threadIdx.x;
    float val = 0.0f;
    if (n < count) {
        val = h[(size_t)bn * DD + d];
        if (n > 0)         val += h[(size_t)(bn - 1) * DD + d];
        if (n + 1 < count) val += h[(size_t)(bn + 1) * DD + d];
        float deg = 1.0f + (n > 0 ? 1.0f : 0.0f) + (n + 1 < count ? 1.0f : 0.0f);
        val /= deg;
    }
    g[(size_t)bn * DD + d] = val;
}

// ---------- readout sum over nodes: hs[b] = sum_{n<count} h[b][n] ----------
__global__ void sumnodes_kernel(const float* __restrict__ h, float* __restrict__ hs, int count) {
    int b = blockIdx.x, d = threadIdx.x;
    float acc = 0.0f;
    for (int n = 0; n < count; ++n) acc += h[((size_t)b * NN + n) * DD + d];
    hs[(size_t)b * DD + d] = acc;
}

__global__ void zero_kernel(float* p, int n) {
    int i = blockIdx.x * blockDim.x + threadIdx.x;
    if (i < n) p[i] = 0.0f;
}
__global__ void copy_kernel(float* dst, const float* src, int n) {
    int i = blockIdx.x * blockDim.x + threadIdx.x;
    if (i < n) dst[i] = src[i];
}
__global__ void writeout_kernel(const float* loss, float* out) { out[0] = loss[0]; }

// =====================================================================================
extern "C" void kernel_launch(void* const* d_in, const int* in_sizes, int n_in,
                              void* d_out, int out_size, void* d_ws, size_t ws_size,
                              hipStream_t stream) {
    (void)in_sizes; (void)n_in; (void)out_size; (void)ws_size;

    const float* hG0    = (const float*)d_in[0];
    const float* W_gcn  = (const float*)d_in[1];
    const float* b_gcn  = (const float*)d_in[2];
    const float* Wg     = (const float*)d_in[3];
    const float* bg     = (const float*)d_in[4];
    const float* fan_W1 = (const float*)d_in[5];
    const float* fan_b1 = (const float*)d_in[6];
    const float* fan_W2 = (const float*)d_in[7];
    const float* fan_b2 = (const float*)d_in[8];
    const float* fae_W1 = (const float*)d_in[9];
    const float* fae_b1 = (const float*)d_in[10];
    const float* fae_W2 = (const float*)d_in[11];
    const float* fae_b2 = (const float*)d_in[12];
    const float* fs_W1  = (const float*)d_in[13];
    const float* fs_b1  = (const float*)d_in[14];
    const float* fs_W2  = (const float*)d_in[15];
    const float* fs_b2  = (const float*)d_in[16];
    const float* fi_W1  = (const float*)d_in[17];
    const float* fi_b1  = (const float*)d_in[18];
    const float* fi_W2  = (const float*)d_in[19];
    const float* fi_b2  = (const float*)d_in[20];

    // ---- bump allocator over workspace (256B aligned) ----
    size_t off = 0;
    auto alloc = [&](size_t bytes) -> void* {
        off = (off + 255) & ~(size_t)255;
        void* p = (char*)d_ws + off;
        off += bytes;
        return p;
    };

    u16* fanW1t = (u16*)alloc(256 * 256 * 2);
    u16* fiW1t  = (u16*)alloc(256 * 256 * 2);
    u16* fiW2t  = (u16*)alloc(256 * 256 * 2);
    u16* Wgt    = (u16*)alloc(256 * 256 * 2);
    u16* gcnW0t = (u16*)alloc(256 * 256 * 2);
    u16* gcnW1t = (u16*)alloc(256 * 256 * 2);
    u16* faeW1t = (u16*)alloc(512 * 512 * 2);
    u16* fsW1t  = (u16*)alloc(512 * 512 * 2);

    float* hG     = (float*)alloc((size_t)BB * 256 * 4);
    float* h      = (float*)alloc((size_t)BB * NN * DD * 4);
    float* g      = (float*)alloc((size_t)BB * NN * DD * 4);
    float* hs     = (float*)alloc((size_t)BB * 256 * 4);
    float* hid1   = (float*)alloc((size_t)BB * 512 * 4);
    float* logits = (float*)alloc((size_t)BB * 2 * 4);
    float* zbuf   = (float*)alloc((size_t)BB * 4);
    float* hidS   = (float*)alloc((size_t)BB * (NN - 1) * 512 * 4);
    float* sbuf   = (float*)alloc((size_t)BB * (NN - 1) * 4);
    float* loss   = (float*)alloc(4);

    // ---- weight conversion (once per call; deterministic) ----
    auto convT = [&](const float* W, u16* Wt, int K, int N) {
        dim3 gr(K / 16, N / 16);
        convT_kernel<<<gr, dim3(16, 16), 0, stream>>>(W, Wt, K, N);
    };
    convT(fan_W1, fanW1t, 256, 256);
    convT(fi_W1,  fiW1t,  256, 256);
    convT(fi_W2,  fiW2t,  256, 256);
    convT(Wg,     Wgt,    256, 256);
    convT(W_gcn,             gcnW0t, 256, 256);
    convT(W_gcn + 256 * 256, gcnW1t, 256, 256);
    convT(fae_W1, faeW1t, 512, 512);
    convT(fs_W1,  fsW1t,  512, 512);

    // ---- init ----
    zero_kernel<<<(BB * NN * DD + 255) / 256, 256, 0, stream>>>(h, BB * NN * DD);
    zero_kernel<<<1, 256, 0, stream>>>(loss, 1);
    copy_kernel<<<(BB * 256 + 255) / 256, 256, 0, stream>>>(hG, hG0, BB * 256);

    auto gemm = [&](const float* A, const float* Asec, int amode, int vv,
                    int M, int N, int K, const u16* Wt,
                    const float* bias, float bscale, float* C, int ldc, int act, int count) {
        dim3 gr(M / 64, N / 64);
        gemm_wmma_kernel<<<gr, 256, 0, stream>>>(A, Asec, amode, vv, Wt, bias, bscale,
                                                 C, ldc, M, N, K, act, count);
    };
    auto head = [&](const float* hid, const float* W2, const float* b2, float* out,
                    int M, int K, int N2) {
        head_kernel<<<(M + 3) / 4, 128, 0, stream>>>(hid, W2, b2, out, M, K, N2);
    };

    // ---- main sequential loop over nodes ----
    for (int v = 0; v < NN; ++v) {
        // add-node head: loss -= mean(log_softmax(fan(hG))[:,1])
        gemm(hG, nullptr, 0, 0, BB, 256, 256, fanW1t, fan_b1, 1.0f, hid1, 256, 1, 0);
        head(hid1, fan_W2, fan_b2, logits, BB, 256, 2);
        loss_bin2_kernel<<<1, 128, 0, stream>>>(logits, 1, loss);

        // hv = fi(hG); h[:,v,:] = hv (written strided into h)
        gemm(hG, nullptr, 0, 0, BB, 256, 256, fiW1t, fi_b1, 1.0f, hid1, 256, 1, 0);
        gemm(hid1, nullptr, 0, 0, BB, 256, 256, fiW2t, fi_b2, 1.0f, h + (size_t)v * DD, NN * DD, 0, 0);

        if (v >= 1) {
            // add-edge yes: z = fae([hG, hv]) (concat fused into GEMM A-stage)
            gemm(hG, h, 2, v, BB, 512, 512, faeW1t, fae_b1, 1.0f, hid1, 512, 1, 0);
            head(hid1, fae_W2, fae_b2, zbuf, BB, 512, 1);
            loss_logsig_kernel<<<1, 128, 0, stream>>>(zbuf, 1.0f, loss);

            // edge-destination scoring: feat gather fused into GEMM A-stage
            gemm(h, nullptr, 1, v, BB * v, 512, 512, fsW1t, fs_b1, 1.0f, hidS, 512, 1, 0);
            head(hidS, fs_W2, fs_b2, sbuf, BB * v, 512, 1);
            loss_rowsm_kernel<<<(BB + 3) / 4, 128, 0, stream>>>(sbuf, v, v - 1, loss);

            // propagate (chain graph, count = v+1), T = 2
            int count = v + 1;
            gather_kernel<<<BB * NN, 256, 0, stream>>>(h, g, count);
            gemm(g, nullptr, 0, 0, BB * NN, 256, 256, gcnW0t, b_gcn, 1.0f, h, DD, 2, count);
            gather_kernel<<<BB * NN, 256, 0, stream>>>(h, g, count);
            gemm(g, nullptr, 0, 0, BB * NN, 256, 256, gcnW1t, b_gcn + 256, 1.0f, h, DD, 2, count);

            // readout: hG = (sum_n h) @ Wg + count*bg
            sumnodes_kernel<<<BB, 256, 0, stream>>>(h, hs, count);
            gemm(hs, nullptr, 0, 0, BB, 256, 256, Wgt, bg, (float)count, hG, 256, 0, 0);
        }

        // add-edge no: z = fae([hG, hv]); loss -= mean(log_sigmoid(-z))
        gemm(hG, h, 2, v, BB, 512, 512, faeW1t, fae_b1, 1.0f, hid1, 512, 1, 0);
        head(hid1, fae_W2, fae_b2, zbuf, BB, 512, 1);
        loss_logsig_kernel<<<1, 128, 0, stream>>>(zbuf, -1.0f, loss);
    }

    // final stop token: loss -= mean(log_softmax(fan(hG))[:,0])
    gemm(hG, nullptr, 0, 0, BB, 256, 256, fanW1t, fan_b1, 1.0f, hid1, 256, 1, 0);
    head(hid1, fan_W2, fan_b2, logits, BB, 256, 2);
    loss_bin2_kernel<<<1, 128, 0, stream>>>(logits, 0, loss);

    writeout_kernel<<<1, 1, 0, stream>>>(loss, (float*)d_out);
}
